// TransformerClassifContact_89051851915578
// MI455X (gfx1250) — matrix operand order repaired and checked by reading, not compiled, not run on hardware
//
#include <hip/hip_runtime.h>
#include <hip/hip_bf16.h>

// MI455X / gfx1250, wave32. bf16 WMMA: v_wmma_f32_16x16x32_bf16.
// k-loops are software-pipelined; __builtin_amdgcn_sched_barrier(0) pins the
// next-iteration global_load_b128 clause above the current WMMA group so the
// wait-count pass emits partial s_wait_loadcnt (overlap) instead of 0.
// Tile sizes chosen for ~2 waves/SIMD occupancy (2048/1024 waves per kernel).

typedef __attribute__((ext_vector_type(16))) __bf16 v16bf;
typedef __attribute__((ext_vector_type(8)))  __bf16 v8bf;
typedef __attribute__((ext_vector_type(8)))  float  v8f;

#define B_  4
#define N_  512
#define D_  1024
#define H_  256
#define M1_ (B_ * N_)   // 2048 rows in stage-1 GEMM

#if defined(__has_builtin) && __has_builtin(__builtin_amdgcn_sched_barrier)
#define SCHED_FENCE() __builtin_amdgcn_sched_barrier(0)
#else
#define SCHED_FENCE()
#endif

// Load a 16-element bf16 WMMA operand as two contiguous 16-byte chunks.
__device__ __forceinline__ v16bf load_bf16_tile(const __bf16* c0, const __bf16* c1) {
    v8bf lo = *reinterpret_cast<const v8bf*>(c0);
    v8bf hi = *reinterpret_cast<const v8bf*>(c1);
    v16bf r;
#pragma unroll
    for (int i = 0; i < 8; ++i) { r[i] = lo[i]; r[i + 8] = hi[i]; }
    return r;
}

__device__ __forceinline__ v8f wmma_bf16(v16bf a, v16bf b, v8f c) {
    return __builtin_amdgcn_wmma_f32_16x16x32_bf16(
        false, a, false, b, (short)0, c, false, false);
}

// ---------------- Prep: f32 -> bf16 for x and W1 ----------------
__global__ __launch_bounds__(256) void prep_bf16_kernel(
    const float* __restrict__ x, const float* __restrict__ w1,
    __bf16* __restrict__ xbf, __bf16* __restrict__ w1bf)
{
    const int idx = blockIdx.x * 256 + threadIdx.x;
    const int nx = M1_ * D_;            // 2,097,152
    const int nw = H_ * D_;             //   262,144
    if (idx < nx) {
        xbf[idx] = (__bf16)x[idx];
    } else if (idx < nx + nw) {
        w1bf[idx - nx] = (__bf16)w1[idx - nx];
    }
}

// ---------------- Stage 1: o = x @ W1^T + b1 ; P_c = o * W2[c] ----------------
// One wave -> 16 rows (mt) x 32 cols (2 B tiles), K = 1024 in 32 WMMA steps.
// 1024 waves = 128 blocks x 8 waves.
__global__ __launch_bounds__(256) void gemm1_kernel(
    const __bf16* __restrict__ xbf, const __bf16* __restrict__ w1bf,
    const float* __restrict__ b1, const float* __restrict__ w2,
    __bf16* __restrict__ obf, __bf16* __restrict__ p0, __bf16* __restrict__ p1)
{
    const int lane = threadIdx.x & 31;
    const int wid  = threadIdx.x >> 5;
    const int gw   = blockIdx.x * 8 + wid;   // 0..1023
    const int mt   = gw >> 3;                // 0..127 (row tile)
    const int nb   = (gw & 7) * 32;          // col base
    const int m16  = lane & 15;
    const int g    = lane >> 4;

    const __bf16* arow = xbf + (size_t)(mt * 16 + m16) * D_;
    const __bf16* brow[2];
#pragma unroll
    for (int t = 0; t < 2; ++t)
        brow[t] = w1bf + (size_t)(nb + t * 16 + m16) * D_;

    v8f acc[2] = {};

    // Pipeline prologue: fragments for kt = 0.
    v16bf a_c = load_bf16_tile(arow + 8 * g, arow + 16 + 8 * g);
    v16bf b_c[2];
#pragma unroll
    for (int t = 0; t < 2; ++t)
        b_c[t] = load_bf16_tile(brow[t] + 8 * g, brow[t] + 16 + 8 * g);

#pragma unroll 2
    for (int kt = 0; kt < D_ / 32 - 1; ++kt) {
        const int kn = (kt + 1) * 32;
        // Prefetch next k-step, then fence so these stay above the WMMAs.
        v16bf a_n = load_bf16_tile(arow + kn + 8 * g, arow + kn + 16 + 8 * g);
        v16bf b_n[2];
#pragma unroll
        for (int t = 0; t < 2; ++t)
            b_n[t] = load_bf16_tile(brow[t] + kn + 8 * g, brow[t] + kn + 16 + 8 * g);
        SCHED_FENCE();
#pragma unroll
        for (int t = 0; t < 2; ++t)
            acc[t] = wmma_bf16(a_c, b_c[t], acc[t]);
        SCHED_FENCE();
        a_c = a_n;
#pragma unroll
        for (int t = 0; t < 2; ++t) b_c[t] = b_n[t];
    }
#pragma unroll
    for (int t = 0; t < 2; ++t)
        acc[t] = wmma_bf16(a_c, b_c[t], acc[t]);

    // Epilogue: +b1, then form P0/P1 in f32 and round to bf16.
#pragma unroll
    for (int t = 0; t < 2; ++t) {
        const int col = nb + t * 16 + m16;   // fixed per lane
        const float bb  = b1[col];
        const float w20 = w2[col];
        const float w21 = w2[H_ + col];
#pragma unroll
        for (int r = 0; r < 8; ++r) {
            const int row = mt * 16 + r + 8 * g;
            const float v = acc[t][r] + bb;
            obf[(size_t)row * H_ + col] = (__bf16)v;
            p0 [(size_t)row * H_ + col] = (__bf16)(v * w20);
            p1 [(size_t)row * H_ + col] = (__bf16)(v * w21);
        }
    }
}

// ---------------- Stage 2: out[b,i,j,c] = P_c[b,i,:] . o[b,j,:] + b2[c] ----------------
// One wave -> one batch b, 16 i-rows (mt), 32 j-cols (2 B tiles), BOTH channels c
// (B tiles shared across c). K = 256 in 8 fully-unrolled, double-buffered steps.
// 2048 waves = 256 blocks x 8 waves.
__global__ __launch_bounds__(256) void gemm2_kernel(
    const __bf16* __restrict__ obf, const __bf16* __restrict__ p0,
    const __bf16* __restrict__ p1, const float* __restrict__ b2,
    float* __restrict__ out)
{
    const int lane = threadIdx.x & 31;
    const int wid  = threadIdx.x >> 5;
    const int gw   = blockIdx.x * 8 + wid;   // 0..2047
    const int b    = gw >> 9;                // 0..3
    const int rem  = gw & 511;
    const int mt   = rem >> 4;               // 0..31 (i tile)
    const int jb   = (rem & 15) * 32;        // j base
    const int m16  = lane & 15;
    const int g    = lane >> 4;

    const __bf16* a0r = p0 + (size_t)b * N_ * H_ + (size_t)(mt * 16 + m16) * H_;
    const __bf16* a1r = p1 + (size_t)b * N_ * H_ + (size_t)(mt * 16 + m16) * H_;
    const __bf16* brow[2];
#pragma unroll
    for (int t = 0; t < 2; ++t)
        brow[t] = obf + (size_t)b * N_ * H_ + (size_t)(jb + t * 16 + m16) * H_;

    v8f acc0[2] = {};
    v8f acc1[2] = {};

    // Prologue: fragments for kt = 0.
    v16bf a0c = load_bf16_tile(a0r + 8 * g, a0r + 16 + 8 * g);
    v16bf a1c = load_bf16_tile(a1r + 8 * g, a1r + 16 + 8 * g);
    v16bf bc[2];
#pragma unroll
    for (int t = 0; t < 2; ++t)
        bc[t] = load_bf16_tile(brow[t] + 8 * g, brow[t] + 16 + 8 * g);

#pragma unroll
    for (int kt = 0; kt < H_ / 32; ++kt) {
        v16bf a0n, a1n, bn[2];
        if (kt < H_ / 32 - 1) {               // static (loop fully unrolled)
            const int kn = (kt + 1) * 32;
            a0n = load_bf16_tile(a0r + kn + 8 * g, a0r + kn + 16 + 8 * g);
            a1n = load_bf16_tile(a1r + kn + 8 * g, a1r + kn + 16 + 8 * g);
#pragma unroll
            for (int t = 0; t < 2; ++t)
                bn[t] = load_bf16_tile(brow[t] + kn + 8 * g, brow[t] + kn + 16 + 8 * g);
            SCHED_FENCE();
        }
#pragma unroll
        for (int t = 0; t < 2; ++t) {
            acc0[t] = wmma_bf16(a0c, bc[t], acc0[t]);
            acc1[t] = wmma_bf16(a1c, bc[t], acc1[t]);
        }
        if (kt < H_ / 32 - 1) {
            SCHED_FENCE();
            a0c = a0n; a1c = a1n;
#pragma unroll
            for (int t = 0; t < 2; ++t) bc[t] = bn[t];
        }
    }

    const float b20 = b2[0];
    const float b21 = b2[1];
#pragma unroll
    for (int t = 0; t < 2; ++t) {
        const int j = jb + t * 16 + m16;
#pragma unroll
        for (int r = 0; r < 8; ++r) {
            const int i = mt * 16 + r + 8 * g;
            float2 v;
            v.x = acc0[t][r] + b20;
            v.y = acc1[t][r] + b21;
            // out layout [b, i, j, c], c innermost -> packed 8-byte store
            *reinterpret_cast<float2*>(
                out + ((((size_t)b * N_ + i) * N_ + j) << 1)) = v;
        }
    }
}

extern "C" void kernel_launch(void* const* d_in, const int* in_sizes, int n_in,
                              void* d_out, int out_size, void* d_ws, size_t ws_size,
                              hipStream_t stream) {
    const float* x  = (const float*)d_in[0];   // [4,512,1024]
    const float* W1 = (const float*)d_in[1];   // [256,1024]
    const float* b1 = (const float*)d_in[2];   // [256]
    const float* W2 = (const float*)d_in[3];   // [2,256]
    const float* b2 = (const float*)d_in[4];   // [2]
    float* out = (float*)d_out;                // [4,512,512,2]

    // Workspace layout (all 16B aligned): 7.5 MB total
    char* ws = (char*)d_ws;
    __bf16* xbf  = (__bf16*)(ws);                                   // 4.0 MB
    __bf16* w1bf = (__bf16*)(ws + (size_t)M1_ * D_ * 2);            // 0.5 MB
    __bf16* obf  = (__bf16*)(ws + (size_t)(M1_ + H_) * D_ * 2);     // 1.0 MB
    __bf16* p0   = obf + (size_t)M1_ * H_;                          // 1.0 MB
    __bf16* p1   = p0  + (size_t)M1_ * H_;                          // 1.0 MB

    const int n_prep = M1_ * D_ + H_ * D_;     // 2,359,296 elements
    prep_bf16_kernel<<<(n_prep + 255) / 256, 256, 0, stream>>>(x, W1, xbf, w1bf);
    gemm1_kernel<<<128, 256, 0, stream>>>(xbf, w1bf, b1, W2, obf, p0, p1);
    gemm2_kernel<<<256, 256, 0, stream>>>(obf, p0, p1, b2, out);
}
